// FPMatMulMESO4D_8194797601026
// MI455X (gfx1250) — matrix operand order repaired and checked by reading, not compiled, not run on hardware
//
#include <hip/hip_runtime.h>

// ---------------------------------------------------------------------------
// Q @ K^T per head, bf16 WMMA path for gfx1250 (MI455X).
// B=1, H=8, L=4096, D=64. Output (1,8,4096,4096) f32.
// One 256-thread block (8 wave32) computes a 128x128 output tile of one head.
// ---------------------------------------------------------------------------

typedef __attribute__((ext_vector_type(16))) __bf16 v16bf;
typedef __attribute__((ext_vector_type(8)))  __bf16 v8bf;
typedef __attribute__((ext_vector_type(8)))  float  v8f;
typedef __attribute__((ext_vector_type(4)))  float  v4f;

#define L_DIM 4096
#define D_DIM 64
#define H_DIM 8
#define TILE  128
#define QS_STRIDE 72    // 64 + 8 bf16 pad  (row stride 144 B, 16B aligned)
#define KT_STRIDE 136   // 128 + 8 bf16 pad (row stride 272 B, 16B aligned)

__global__ __launch_bounds__(256)
void qkT_wmma_bf16_kernel(const float* __restrict__ Q,
                          const float* __restrict__ K,
                          float* __restrict__ Out)
{
    __shared__ __bf16 Qs[TILE * QS_STRIDE];   // [q_row][d]   row-major
    __shared__ __bf16 Kt[D_DIM * KT_STRIDE];  // [d][k_row]   transposed

    const int h       = blockIdx.z;
    const int rowTile = blockIdx.y * TILE;    // query rows
    const int colTile = blockIdx.x * TILE;    // key rows (output cols)

    const float* Qg = Q + ((size_t)h * L_DIM + rowTile) * D_DIM;
    const float* Kg = K + ((size_t)h * L_DIM + colTile) * D_DIM;
    float*       Og = Out + (size_t)h * L_DIM * L_DIM;

    const int tid = threadIdx.x;              // 0..255

    // ---- global -> LDS, f32 -> bf16; Q row-major, K transposed ------------
    // 128 rows x 64 cols = 2048 float4 per tile; 8 float4 per thread.
#pragma unroll
    for (int it = 0; it < 8; ++it) {
        int i   = tid + it * 256;             // float4 index 0..2047
        int row = i >> 4;                     // 16 float4 per 64-elem row
        int c4  = i & 15;

        v4f q = *(const v4f*)(Qg + row * D_DIM + c4 * 4);
        v4f k = *(const v4f*)(Kg + row * D_DIM + c4 * 4);

        __bf16* qdst = &Qs[row * QS_STRIDE + c4 * 4];
        qdst[0] = (__bf16)q.x; qdst[1] = (__bf16)q.y;
        qdst[2] = (__bf16)q.z; qdst[3] = (__bf16)q.w;

        Kt[(c4 * 4 + 0) * KT_STRIDE + row] = (__bf16)k.x;
        Kt[(c4 * 4 + 1) * KT_STRIDE + row] = (__bf16)k.y;
        Kt[(c4 * 4 + 2) * KT_STRIDE + row] = (__bf16)k.z;
        Kt[(c4 * 4 + 3) * KT_STRIDE + row] = (__bf16)k.w;
    }
    __syncthreads();

    // ---- per-wave 32x64 sub-tile: 2x4 WMMA tiles --------------------------
    const int wave  = tid >> 5;               // 0..7
    const int lane  = tid & 31;
    const int mBase = (wave >> 1) * 32;       // 0,32,64,96
    const int nBase = (wave & 1) * 64;        // 0,64

    v8f acc[2][4] = {};

#pragma unroll
    for (int d0 = 0; d0 < D_DIM; d0 += 32) {
        // A fragments (16x32 bf16): lanes 0-15 hold row M=lane with
        // K in {0..7, 16..23}; lanes 16-31 hold same rows, K in {8..15, 24..31}.
        v16bf afr[2];
#pragma unroll
        for (int t = 0; t < 2; ++t) {
            int row = mBase + t * 16 + (lane & 15);
            int db  = d0 + ((lane >> 4) << 3);            // +0 or +8
            const __bf16* src = &Qs[row * QS_STRIDE + db];
            v8bf lo = *(const v8bf*)(src);                // K = db .. db+7
            v8bf hi = *(const v8bf*)(src + 16);           // K = db+16 .. db+23
            v16bf a;
#pragma unroll
            for (int e = 0; e < 8; ++e) { a[e] = lo[e]; a[8 + e] = hi[e]; }
            afr[t] = a;
        }

        // B fragments (32x16 bf16): lane l holds column N = l&15, K in
        // {0..7, 16..23} (lanes 0-15) or {8..15, 24..31} (lanes 16-31).
        v16bf bfr[4];
#pragma unroll
        for (int t = 0; t < 4; ++t) {
            const __bf16* src = &Kt[(d0 + ((lane >> 4) << 3)) * KT_STRIDE + nBase + t * 16 + (lane & 15)];
            v16bf b;
#pragma unroll
            for (int e = 0; e < 8; ++e) { b[e] = src[e * KT_STRIDE]; b[8 + e] = src[(16 + e) * KT_STRIDE]; }
            bfr[t] = b;
        }

#pragma unroll
        for (int mt = 0; mt < 2; ++mt) {
#pragma unroll
            for (int nt = 0; nt < 4; ++nt) {
                // (neg_a, A, neg_b, B, c_mod, C, reuse_a, reuse_b)
                acc[mt][nt] = __builtin_amdgcn_wmma_f32_16x16x32_bf16(
                    false, afr[mt], false, bfr[nt],
                    (short)0, acc[mt][nt], false, false);
            }
        }
    }

    // ---- store: VGPR r -> (M = r + 8*(lane>=16), N = lane&15) -------------
#pragma unroll
    for (int mt = 0; mt < 2; ++mt) {
#pragma unroll
        for (int nt = 0; nt < 4; ++nt) {
            int col   = colTile + nBase + nt * 16 + (lane & 15);
            int rbase = rowTile + mBase + mt * 16 + ((lane >> 4) << 3);
#pragma unroll
            for (int r = 0; r < 8; ++r) {
                Og[(size_t)(rbase + r) * L_DIM + col] = acc[mt][nt][r];
            }
        }
    }
}

extern "C" void kernel_launch(void* const* d_in, const int* in_sizes, int n_in,
                              void* d_out, int out_size, void* d_ws, size_t ws_size,
                              hipStream_t stream) {
    const float* Q  = (const float*)d_in[0];
    const float* K  = (const float*)d_in[1];
    float*       Ot = (float*)d_out;

    dim3 grid(L_DIM / TILE, L_DIM / TILE, H_DIM);  // 32 x 32 x 8
    dim3 block(256);
    qkT_wmma_bf16_kernel<<<grid, block, 0, stream>>>(Q, K, Ot);
}
